// MultiHeadAttention_5454608466014
// MI455X (gfx1250) — matrix operand order repaired and checked by reading, not compiled, not run on hardware
//
#include <hip/hip_runtime.h>
#include <hip/hip_bf16.h>
#include <math.h>

typedef __attribute__((ext_vector_type(16))) __bf16 v16bf;
typedef __attribute__((ext_vector_type(8)))  float  v8f;

union Frag { v16bf v; uint4 q[2]; };

__device__ __forceinline__ unsigned short f2bf(float f) {
    unsigned int u = __float_as_uint(f);
    unsigned int r = u + 0x7FFFu + ((u >> 16) & 1u);
    return (unsigned short)(r >> 16);
}

// ---------------------------------------------------------------------------
// Tensor Data Mover: 2D tile load Global -> LDS (bf16 elements).
// D# built per CDNA5 ISA sec 8.3-8.5; issued by one wave, completion via
// s_wait_tensorcnt. Falls back to plain vector copies if builtin is absent.
// ---------------------------------------------------------------------------
#if __has_builtin(__builtin_amdgcn_tensor_load_to_lds) && \
    __has_builtin(__builtin_amdgcn_s_wait_tensorcnt)
#define HAVE_TDM 1
#if __has_include(<hip/amd_detail/amd_gfx1250_TDM.h>)
#define TDM_SIX_ARGS 1
#endif
typedef __attribute__((ext_vector_type(4))) unsigned int u32x4;
typedef __attribute__((ext_vector_type(4))) int i32x4;
typedef __attribute__((ext_vector_type(8))) int i32x8;

__device__ __forceinline__ void tdm_load_2d_bf16(
    unsigned lds_off, const void* gaddr,
    unsigned tile_d0, unsigned tile_d1,        // tile size in elements
    unsigned tensor_d0, unsigned tensor_d1,    // tensor extent in elements
    unsigned stride_d0)                        // row stride in elements
{
    unsigned long long ga = (unsigned long long)(size_t)gaddr;
    u32x4 g0;
    g0.x = 1u;                                     // count=1, user mode
    g0.y = lds_off;                                // lds_addr (bytes)
    g0.z = (unsigned)(ga & 0xFFFFFFFFu);           // global_addr[31:0]
    g0.w = (unsigned)((ga >> 32) & 0x01FFFFFFu) | (2u << 30); // addr[56:32]|type=2
    i32x8 g1;
    g1[0] = (int)(1u << 16);                                   // data_size=2B
    g1[1] = (int)((tensor_d0 & 0xFFFFu) << 16);                // dim0[15:0]
    g1[2] = (int)((tensor_d0 >> 16) | ((tensor_d1 & 0xFFFFu) << 16));
    g1[3] = (int)((tensor_d1 >> 16) | (tile_d0 << 16));        // dim1 hi|tile0
    g1[4] = (int)(tile_d1 & 0xFFFFu);                          // tile1, tile2=0
    g1[5] = (int)stride_d0;                                    // dim0_stride lo
    g1[6] = 0;
    g1[7] = 0;
    i32x4 z4 = {0, 0, 0, 0};
#ifdef TDM_SIX_ARGS
    i32x8 z8 = {0, 0, 0, 0, 0, 0, 0, 0};
    __builtin_amdgcn_tensor_load_to_lds(g0, g1, z4, z4, z8, 0);
#else
    __builtin_amdgcn_tensor_load_to_lds(g0, g1, z4, z4, 0);
#endif
}
#endif

__global__ void f32_to_bf16_kernel(const float* __restrict__ in,
                                   unsigned short* __restrict__ out, int n) {
    for (int i = blockIdx.x * blockDim.x + threadIdx.x; i < n;
         i += gridDim.x * blockDim.x)
        out[i] = f2bf(in[i]);
}

// ---------------------------------------------------------------------------
// Tiled bf16 WMMA GEMM: C[M,N] = A[M,K] * B[K,N] + bias
//  block = 256 threads (8 waves), block tile 128x128, wave tile 32x64
//  MODE 0: store fp32 (row-major) | MODE 1: RoPE->bf16 (B,H,S,Ld)
//  MODE 2: plain bf16 (B,H,S,Ld)
// ---------------------------------------------------------------------------
template <int MODE, int Hc, int Ldc, int Sc>
__global__ __launch_bounds__(256, 1) void gemm_bf16_wmma(
    const unsigned short* __restrict__ A,
    const unsigned short* __restrict__ Bw,
    const float* __restrict__ bias,
    float* __restrict__ Cf,
    unsigned short* __restrict__ Cbf,
    int M, int N, int K)
{
    __shared__ __align__(16) unsigned short Al[128 * 32];   // [m][k]
    __shared__ __align__(16) unsigned short Bl[128 * 32];   // [n][k] transposed

    const int t    = threadIdx.x;
    const int lane = t & 31;
    const int w    = t >> 5;
    const int half = lane >> 4;
    const int l16  = lane & 15;
    const int m0   = blockIdx.y * 128;
    const int n0   = blockIdx.x * 128;
    const int wm   = w >> 1;
    const int wn   = w & 1;

    const v8f vzero = {0.f, 0.f, 0.f, 0.f, 0.f, 0.f, 0.f, 0.f};
    v8f acc[2][4];
#pragma unroll
    for (int i = 0; i < 2; ++i)
#pragma unroll
        for (int j = 0; j < 4; ++j) acc[i][j] = vzero;

    for (int k0 = 0; k0 < K; k0 += 32) {
#ifdef HAVE_TDM
        // A tile via Tensor Data Mover (row-major 128x32 copy), wave 0 only.
        if (w == 0) {
            tdm_load_2d_bf16((unsigned)(size_t)&Al[0],
                             &A[(size_t)m0 * K + k0],
                             32u, 128u, 32u, 128u, (unsigned)K);
        }
#else
#pragma unroll
        for (int i = 0; i < 2; ++i) {
            int c   = i * 256 + t;
            int row = c >> 2;
            int kc  = (c & 3) << 3;
            *(uint4*)(&Al[row * 32 + kc]) =
                *(const uint4*)(&A[(size_t)(m0 + row) * K + k0 + kc]);
        }
#endif
        // B tile transposed: Bl[n][k] (manual — TDM cannot transpose)
#pragma unroll
        for (int i = 0; i < 2; ++i) {
            int c  = i * 256 + t;
            int kr = c >> 4;
            int nc = (c & 15) << 3;
            union { uint4 q; unsigned short s[8]; } u;
            u.q = *(const uint4*)(&Bw[(size_t)(k0 + kr) * N + n0 + nc]);
#pragma unroll
            for (int j = 0; j < 8; ++j) Bl[(nc + j) * 32 + kr] = u.s[j];
        }
#ifdef HAVE_TDM
        if (w == 0) __builtin_amdgcn_s_wait_tensorcnt(0);
#endif
        __syncthreads();

        Frag af[2], bf[4];
#pragma unroll
        for (int i = 0; i < 2; ++i) {
            int row = wm * 32 + i * 16 + l16;
            af[i].q[0] = *(const uint4*)(&Al[row * 32 + half * 8]);
            af[i].q[1] = *(const uint4*)(&Al[row * 32 + 16 + half * 8]);
        }
#pragma unroll
        for (int j = 0; j < 4; ++j) {
            int col = wn * 64 + j * 16 + l16;
            bf[j].q[0] = *(const uint4*)(&Bl[col * 32 + half * 8]);
            bf[j].q[1] = *(const uint4*)(&Bl[col * 32 + 16 + half * 8]);
        }
#pragma unroll
        for (int i = 0; i < 2; ++i)
#pragma unroll
            for (int j = 0; j < 4; ++j)
                acc[i][j] = __builtin_amdgcn_wmma_f32_16x16x32_bf16(
                    false, af[i].v, false, bf[j].v, (short)0, acc[i][j],
                    false, false);
        __syncthreads();
    }

    // epilogue (all div/mod by compile-time powers of two)
#pragma unroll
    for (int j = 0; j < 4; ++j) {
        int gcol = n0 + wn * 64 + j * 16 + l16;
        float bj = bias[gcol];
#pragma unroll
        for (int i = 0; i < 2; ++i) {
#pragma unroll
            for (int r = 0; r < 8; ++r) {
                int grow = m0 + wm * 32 + i * 16 + r + half * 8;
                float val = acc[i][j][r] + bj;
                if constexpr (MODE == 0) {
                    Cf[(size_t)grow * N + gcol] = val;
                } else {
                    int bb = grow / Sc, s = grow % Sc;
                    int hh = gcol / Ldc, c = gcol % Ldc;
                    size_t hbase = (((size_t)(bb * Hc + hh) * Sc + s) * Ldc);
                    if constexpr (MODE == 2) {
                        Cbf[hbase + c] = f2bf(val);
                    } else {
                        // RoPE: pair (2*j2, 2*j2+1) lives in adjacent lanes
                        float partner = __shfl_xor(val, 1);
                        int j2 = c >> 1;
                        float inv_freq =
                            __expf(-(2.0f * (float)j2 / (float)Ldc) *
                                   9.210340371976184f);   // ln(10000)
                        float ang = (float)s * inv_freq;
                        float sn, cs;
                        __sincosf(ang, &sn, &cs);
                        float outv; int d;
                        if ((c & 1) == 0) { outv = val * cs - partner * sn; d = j2; }
                        else { outv = partner * sn + val * cs; d = (Ldc >> 1) + j2; }
                        Cbf[hbase + d] = f2bf(outv);
                    }
                }
            }
        }
    }
}

// ---------------------------------------------------------------------------
// Flash attention (causal + query mask), bf16 WMMA, online softmax.
// One block = (b, h, 128 query rows); 8 waves x 16 q-rows; key tiles of 64.
// ---------------------------------------------------------------------------
template <int Bc, int Hc, int Sc, int Ldc>
__global__ __launch_bounds__(256, 1) void flash_attn_wmma(
    const unsigned short* __restrict__ Q,
    const unsigned short* __restrict__ Kr,
    const unsigned short* __restrict__ Vr,
    const int* __restrict__ mask,
    unsigned short* __restrict__ Oo)
{
    __shared__ __align__(16) unsigned short Kl[64 * Ldc];   // [key][ld]
    __shared__ __align__(16) unsigned short Vt[Ldc * 64];   // [ld][key]
    __shared__ __align__(16) unsigned short Pl[8][16 * 64]; // per-wave P

    const int t    = threadIdx.x;
    const int lane = t & 31;
    const int w    = t >> 5;
    const int half = lane >> 4;
    const int l16  = lane & 15;

    constexpr int nqt = Sc / 128;
    const int qt  = blockIdx.x % nqt;
    const int h   = (blockIdx.x / nqt) % Hc;
    const int b   = blockIdx.x / (nqt * Hc);
    const int q0  = qt * 128;
    const size_t hb = (size_t)(b * Hc + h) * (size_t)Sc * (size_t)Ldc;
    const int qrow0 = q0 + w * 16;

    // preload Q A-fragments (16 rows x full Ld = 4 k-steps)
    Frag qf[4];
    {
        const unsigned short* qp = Q + hb + (size_t)(qrow0 + l16) * Ldc;
#pragma unroll
        for (int kk = 0; kk < 4; ++kk) {
            qf[kk].q[0] = *(const uint4*)(qp + kk * 32 + half * 8);
            qf[kk].q[1] = *(const uint4*)(qp + kk * 32 + 16 + half * 8);
        }
    }

    const v8f vzero = {0.f, 0.f, 0.f, 0.f, 0.f, 0.f, 0.f, 0.f};
    float mrow[8], lrow[8];
    int   qm[8];
    v8f   o[8];
#pragma unroll
    for (int r = 0; r < 8; ++r) {
        mrow[r] = -__builtin_inff();
        lrow[r] = 0.f;
        qm[r]   = mask[b * Sc + qrow0 + r + half * 8];
        o[r]    = vzero;
    }

    const float scale = 0.08838834764831845f;  // 1/sqrt(128)
    const int nkt = q0 / 64 + 2;

    for (int kt = 0; kt < nkt; ++kt) {
        const int k0 = kt * 64;
#ifdef HAVE_TDM
        // K tile (row-major 64 x Ld) via Tensor Data Mover, wave 0 only.
        if (w == 0) {
            tdm_load_2d_bf16((unsigned)(size_t)&Kl[0],
                             &Kr[hb + (size_t)k0 * Ldc],
                             (unsigned)Ldc, 64u, (unsigned)Ldc, 64u,
                             (unsigned)Ldc);
        }
#else
#pragma unroll
        for (int i = 0; i < 4; ++i) {
            int c   = i * 256 + t;
            int row = c >> 4;
            int kc  = (c & 15) << 3;
            *(uint4*)(&Kl[row * Ldc + kc]) =
                *(const uint4*)(&Kr[hb + (size_t)(k0 + row) * Ldc + kc]);
        }
#endif
        // V tile transposed: Vt[ld][key] (manual transpose)
#pragma unroll
        for (int i = 0; i < 4; ++i) {
            int c   = i * 256 + t;
            int key = c >> 4;
            int ldc = (c & 15) << 3;
            union { uint4 q; unsigned short s[8]; } u;
            u.q = *(const uint4*)(&Vr[hb + (size_t)(k0 + key) * Ldc + ldc]);
#pragma unroll
            for (int j = 0; j < 8; ++j) Vt[(ldc + j) * 64 + key] = u.s[j];
        }
#ifdef HAVE_TDM
        if (w == 0) __builtin_amdgcn_s_wait_tensorcnt(0);
#endif
        __syncthreads();

        // S = Q K^T  (16 x 64 per wave)
        v8f sacc[4];
#pragma unroll
        for (int nt = 0; nt < 4; ++nt) {
            sacc[nt] = vzero;
            int kcol = nt * 16 + l16;
#pragma unroll
            for (int kk = 0; kk < 4; ++kk) {
                Frag bfr;
                bfr.q[0] = *(const uint4*)(&Kl[kcol * Ldc + kk * 32 + half * 8]);
                bfr.q[1] = *(const uint4*)(&Kl[kcol * Ldc + kk * 32 + 16 + half * 8]);
                sacc[nt] = __builtin_amdgcn_wmma_f32_16x16x32_bf16(
                    false, qf[kk].v, false, bfr.v, (short)0, sacc[nt],
                    false, false);
            }
        }

        // scale + causal/query mask + online softmax (half-wave reductions)
        float mnew[8];
#pragma unroll
        for (int r = 0; r < 8; ++r) {
            int grow = qrow0 + r + half * 8;
            float mx = mrow[r];
#pragma unroll
            for (int nt = 0; nt < 4; ++nt) {
                int k = k0 + nt * 16 + l16;
                float s = sacc[nt][r] * scale;
                if (k > grow || qm[r] == 0) s = -__builtin_inff();
                sacc[nt][r] = s;
                mx = fmaxf(mx, s);
            }
            mx = fmaxf(mx, __shfl_xor(mx, 1));
            mx = fmaxf(mx, __shfl_xor(mx, 2));
            mx = fmaxf(mx, __shfl_xor(mx, 4));
            mx = fmaxf(mx, __shfl_xor(mx, 8));
            mnew[r] = mx;
        }
#pragma unroll
        for (int r = 0; r < 8; ++r) {
            float alpha = __expf(mrow[r] - mnew[r]);
            mrow[r] = mnew[r];
            float ps = 0.f;
#pragma unroll
            for (int nt = 0; nt < 4; ++nt) {
                float p = __expf(sacc[nt][r] - mnew[r]);
                sacc[nt][r] = p;
                ps += p;
            }
            ps += __shfl_xor(ps, 1);
            ps += __shfl_xor(ps, 2);
            ps += __shfl_xor(ps, 4);
            ps += __shfl_xor(ps, 8);
            lrow[r] = lrow[r] * alpha + ps;
#pragma unroll
            for (int n = 0; n < 8; ++n) o[n][r] *= alpha;
        }

        // P: C-layout regs -> per-wave LDS -> A-fragments
#pragma unroll
        for (int nt = 0; nt < 4; ++nt)
#pragma unroll
            for (int r = 0; r < 8; ++r)
                Pl[w][(r + half * 8) * 64 + nt * 16 + l16] = f2bf(sacc[nt][r]);
        asm volatile("s_wait_dscnt 0" ::: "memory");

        Frag pf[2];
#pragma unroll
        for (int kk2 = 0; kk2 < 2; ++kk2) {
            pf[kk2].q[0] = *(const uint4*)(&Pl[w][l16 * 64 + kk2 * 32 + half * 8]);
            pf[kk2].q[1] = *(const uint4*)(&Pl[w][l16 * 64 + kk2 * 32 + 16 + half * 8]);
        }

        // O += P V
#pragma unroll
        for (int n = 0; n < 8; ++n) {
            int vcol = n * 16 + l16;
#pragma unroll
            for (int kk2 = 0; kk2 < 2; ++kk2) {
                Frag vf;
                vf.q[0] = *(const uint4*)(&Vt[vcol * 64 + kk2 * 32 + half * 8]);
                vf.q[1] = *(const uint4*)(&Vt[vcol * 64 + kk2 * 32 + 16 + half * 8]);
                o[n] = __builtin_amdgcn_wmma_f32_16x16x32_bf16(
                    false, pf[kk2].v, false, vf.v, (short)0, o[n],
                    false, false);
            }
        }
        __syncthreads();
    }

    // epilogue: normalize, store attended (B*S, H*Ld) bf16
#pragma unroll
    for (int r = 0; r < 8; ++r) {
        float inv = 1.0f / lrow[r];
        int grow = qrow0 + r + half * 8;
        size_t rowbase = (size_t)(b * Sc + grow) * (size_t)(Hc * Ldc) +
                         (size_t)h * Ldc;
#pragma unroll
        for (int n = 0; n < 8; ++n)
            Oo[rowbase + n * 16 + l16] = f2bf(o[n][r] * inv);
    }
}

// ---------------------------------------------------------------------------
extern "C" void kernel_launch(void* const* d_in, const int* in_sizes, int n_in,
                              void* d_out, int out_size, void* d_ws, size_t ws_size,
                              hipStream_t stream) {
    const float* x   = (const float*)d_in[0];
    const int*   am  = (const int*)d_in[1];
    const float* Wq  = (const float*)d_in[2];
    const float* bq  = (const float*)d_in[3];
    const float* Wk  = (const float*)d_in[4];
    const float* bk  = (const float*)d_in[5];
    const float* Wv  = (const float*)d_in[6];
    const float* bv  = (const float*)d_in[7];
    const float* Wo  = (const float*)d_in[8];
    const float* bo  = (const float*)d_in[9];

    constexpr int Bb = 2, S = 2048, D = 2048, H = 16, Ld = 128;
    const int M = Bb * S;        // 4096
    const int N = H * Ld;        // 2048
    const int K = D;             // 2048

    char* ws = (char*)d_ws;
    size_t off = 0;
    auto alloc = [&](size_t bytes) -> void* {
        off = (off + 255) & ~(size_t)255;
        void* p = ws + off;
        off += bytes;
        return p;
    };
    unsigned short* xbf = (unsigned short*)alloc((size_t)M * K * 2);
    unsigned short* Wqb = (unsigned short*)alloc((size_t)K * N * 2);
    unsigned short* Wkb = (unsigned short*)alloc((size_t)K * N * 2);
    unsigned short* Wvb = (unsigned short*)alloc((size_t)K * N * 2);
    unsigned short* Wob = (unsigned short*)alloc((size_t)N * D * 2);
    unsigned short* Qr  = (unsigned short*)alloc((size_t)M * N * 2);
    unsigned short* Krr = (unsigned short*)alloc((size_t)M * N * 2);
    unsigned short* Vrr = (unsigned short*)alloc((size_t)M * N * 2);
    unsigned short* att = (unsigned short*)alloc((size_t)M * N * 2);

    f32_to_bf16_kernel<<<2048, 256, 0, stream>>>(x,  xbf, M * K);
    f32_to_bf16_kernel<<<2048, 256, 0, stream>>>(Wq, Wqb, K * N);
    f32_to_bf16_kernel<<<2048, 256, 0, stream>>>(Wk, Wkb, K * N);
    f32_to_bf16_kernel<<<2048, 256, 0, stream>>>(Wv, Wvb, K * N);
    f32_to_bf16_kernel<<<2048, 256, 0, stream>>>(Wo, Wob, N * D);

    dim3 g(N / 128, M / 128);
    gemm_bf16_wmma<1, H, Ld, S><<<g, 256, 0, stream>>>(xbf, Wqb, bq, nullptr,
                                                       Qr, M, N, K);
    gemm_bf16_wmma<1, H, Ld, S><<<g, 256, 0, stream>>>(xbf, Wkb, bk, nullptr,
                                                       Krr, M, N, K);
    gemm_bf16_wmma<2, H, Ld, S><<<g, 256, 0, stream>>>(xbf, Wvb, bv, nullptr,
                                                       Vrr, M, N, K);

    flash_attn_wmma<Bb, H, S, Ld><<<Bb * H * (S / 128), 256, 0, stream>>>(
        Qr, Krr, Vrr, am, att);

    dim3 go(D / 128, M / 128);
    gemm_bf16_wmma<0, H, Ld, S><<<go, 256, 0, stream>>>(att, Wob, bo,
                                                        (float*)d_out, nullptr,
                                                        M, D, N);
}
